// MultiHeadSelfAttention_5901285064785
// MI455X (gfx1250) — matrix operand order repaired and checked by reading, not compiled, not run on hardware
//
#include <hip/hip_runtime.h>

// ---------------------------------------------------------------------------
// MHA for MI455X (gfx1250): bf16 WMMA everywhere, flash-attention middle.
// b=4, s=2048, d_model=1024, h=16, m=64.
// All right-hand (B) matrices are stored N-major so every WMMA fragment is
// loaded with contiguous 16B vector loads (no scalar gather + packing).
// ---------------------------------------------------------------------------

typedef __bf16 bf16_t;
typedef __attribute__((ext_vector_type(16))) __bf16 v16bf;
typedef __attribute__((ext_vector_type(8)))  __bf16 v8bf;
typedef __attribute__((ext_vector_type(8)))  float  v8f;

#define BATCH 4
#define SEQ   2048
#define DMODEL 1024
#define NHEADS 16
#define MHEAD 64

// ---- workspace layout (bytes) ----
#define OFF_XH   (0)                            // x bf16 [4,2048,1024]      = 16 MiB
#define OFF_WQ   (OFF_XH + 16777216)            // W_q^T bf16 [16,64,1024]   = 2 MiB
#define OFF_WK   (OFF_WQ + 2097152)             // W_k^T bf16 [16,64,1024]
#define OFF_WV   (OFF_WK + 2097152)             // W_v^T bf16 [16,64,1024]
#define OFF_WO   (OFF_WV + 2097152)             // W_o^T bf16 [16,1024,64]   = 2 MiB
#define OFF_Q    (OFF_WO + 2097152)             // q bf16 [4,16,2048,64]     = 16 MiB
#define OFF_K    (OFF_Q  + 16777216)            // k bf16 [4,16,2048,64]
#define OFF_V    (OFF_K  + 16777216)            // v^T bf16 [4,16,64,2048]
#define OFF_CTX  (OFF_V  + 16777216)            // ctx bf16 [4,16,2048,64]

// ---------------------------------------------------------------------------
// WMMA fragment loaders (contiguous k order: lane half h holds K = 16h..16h+15, wave32)
// ---------------------------------------------------------------------------

// A-matrix 16x32 bf16, row-major source with stride lda (elements).
// lane 0-15: row=lane,    K = 0..15
// lane16-31: row=lane-16, K = 16..31
static __device__ __forceinline__ v16bf load_a_frag(const bf16_t* base, int lda) {
    const int lane = threadIdx.x & 31;
    const int row  = lane & 15;
    const int kb   = (lane >> 4) << 4;          // 0 or 16
    const bf16_t* p = base + row * lda + kb;
    v8bf lo = *reinterpret_cast<const v8bf*>(p);        // K = kb..kb+7
    v8bf hi = *reinterpret_cast<const v8bf*>(p + 8);    // K = kb+8..kb+15
    v16bf r;
#pragma unroll
    for (int i = 0; i < 8; ++i) { r[i] = lo[i]; r[i + 8] = hi[i]; }
    return r;
}

// B-matrix 32x16 bf16 from an N-major source (each column n holds contiguous K):
// value B[k][n] = base[n*ldn + kofs + k]  -> two contiguous 16B loads per lane.
// lane 0-15: col=lane, K = 0..15 ; lane16-31: col=lane-16, K = 16..31
static __device__ __forceinline__ v16bf load_b_frag_T(const bf16_t* base, int ldn, int kofs) {
    const int lane = threadIdx.x & 31;
    const int col  = lane & 15;
    const int kb   = (lane >> 4) << 4;          // 0 or 16
    const bf16_t* p = base + col * ldn + kofs + kb;
    v8bf lo = *reinterpret_cast<const v8bf*>(p);
    v8bf hi = *reinterpret_cast<const v8bf*>(p + 8);
    v16bf r;
#pragma unroll
    for (int i = 0; i < 8; ++i) { r[i] = lo[i]; r[i + 8] = hi[i]; }
    return r;
}

static __device__ __forceinline__ v8f wmma_bf16(v16bf a, v16bf b, v8f c) {
    return __builtin_amdgcn_wmma_f32_16x16x32_bf16(false, a, false, b, (short)0, c,
                                                   false, false);
}

// ---------------------------------------------------------------------------
// Kernel 1a: plain f32 -> bf16 convert (x)
// ---------------------------------------------------------------------------
__global__ void cvt_f32_bf16(const float* __restrict__ src, bf16_t* __restrict__ dst, int n) {
    int i = blockIdx.x * 256 + threadIdx.x;
    if (i < n) dst[i] = (bf16_t)src[i];
}

// Kernel 1b: W_{q,k,v} [h,d,m] f32 -> transposed bf16 [h,m,d]
__global__ void cvt_tr_qkvw(const float* __restrict__ src, bf16_t* __restrict__ dst) {
    int i = blockIdx.x * 256 + threadIdx.x;     // over out = [h][m][d]
    int h = i >> 16;                            // 64*1024 per head
    int r = i & 65535;
    int m = r >> 10;
    int d = r & 1023;
    dst[i] = (bf16_t)src[(h * DMODEL + d) * MHEAD + m];
}

// Kernel 1c: W_o [h,m,d] f32 -> transposed bf16 [h,d,m]
__global__ void cvt_tr_ow(const float* __restrict__ src, bf16_t* __restrict__ dst) {
    int i = blockIdx.x * 256 + threadIdx.x;     // over out = [h][d][m]
    int h = i >> 16;
    int r = i & 65535;
    int d = r >> 6;
    int m = r & 63;
    dst[i] = (bf16_t)src[(h * MHEAD + m) * DMODEL + d];
}

// ---------------------------------------------------------------------------
// Kernel 2: QKV projection.  grid = (128 s-tiles, 4 batches), 384 thr (12 waves)
// wave w: mat = w/4 (q,k,v), n-tile = w%4 (16 cols of m=64); loops 16 heads.
// Weights are N-major [h,m,d] -> contiguous B fragments.  V stored transposed.
// ---------------------------------------------------------------------------
#define XS_LD 1032   // 1024 + 8 pad: keeps 16B alignment, spreads LDS banks
__global__ __launch_bounds__(384) void qkv_kernel(
    const bf16_t* __restrict__ xh,
    const bf16_t* __restrict__ wqT, const bf16_t* __restrict__ wkT,
    const bf16_t* __restrict__ wvT,
    bf16_t* __restrict__ q, bf16_t* __restrict__ k, bf16_t* __restrict__ vT)
{
    __shared__ bf16_t xs[16 * XS_LD];

    const int s0 = blockIdx.x * 16;
    const int b  = blockIdx.y;

    // cooperative load of the 16x1024 x tile (bf16) into LDS
    const bf16_t* xsrc = xh + (b * SEQ + s0) * DMODEL;
    for (int i = threadIdx.x; i < 1024; i += 384) {   // 1024 chunks of 16 elems
        int r = i >> 6;                 // 64 chunks per row
        int c = (i & 63) << 4;
        *reinterpret_cast<v8bf*>(&xs[r * XS_LD + c])     =
            *reinterpret_cast<const v8bf*>(xsrc + r * DMODEL + c);
        *reinterpret_cast<v8bf*>(&xs[r * XS_LD + c + 8]) =
            *reinterpret_cast<const v8bf*>(xsrc + r * DMODEL + c + 8);
    }
    __syncthreads();

    const int wid  = threadIdx.x >> 5;
    const int mat  = wid >> 2;                  // 0=q 1=k 2=v
    const int nt   = wid & 3;                   // n-tile within m=64
    const int lane = threadIdx.x & 31;
    const int crow = lane >> 4;
    const int ccol = lane & 15;

    const bf16_t* W = (mat == 0) ? wqT : (mat == 1) ? wkT : wvT;
    const float oscale = (mat == 0) ? 0.125f : 1.0f;   // 1/sqrt(64) folded into q

    for (int h = 0; h < NHEADS; ++h) {
        v8f acc = {};
        // W^T layout [h, m, d]: B[k][n] = W^T[(h*64 + nt*16 + n)*1024 + kk*32 + k]
        const bf16_t* Wh = W + (size_t)(h * MHEAD + nt * 16) * DMODEL;
#pragma unroll 4
        for (int kk = 0; kk < DMODEL / 32; ++kk) {
            v16bf a  = load_a_frag(&xs[kk * 32], XS_LD);
            v16bf bb = load_b_frag_T(Wh + kk * 32, DMODEL, 0);
            acc = wmma_bf16(a, bb, acc);
        }
        if (mat == 2) {
            // v stored transposed: vT[b][h][m][s]
            bf16_t* dst = vT + ((size_t)(b * NHEADS + h) * MHEAD) * SEQ;
#pragma unroll
            for (int r = 0; r < 8; ++r) {
                int row = r + crow * 8;                // s-row within tile
                int mi  = nt * 16 + ccol;              // m index
                dst[(size_t)mi * SEQ + s0 + row] = (bf16_t)acc[r];
            }
        } else {
            bf16_t* dstm = (mat == 0) ? q : k;
            bf16_t* dst = dstm + ((size_t)(b * NHEADS + h) * SEQ + s0) * MHEAD + nt * 16;
#pragma unroll
            for (int r = 0; r < 8; ++r) {
                int row = r + crow * 8;
                dst[row * MHEAD + ccol] = (bf16_t)(acc[r] * oscale);
            }
        }
    }
}

// ---------------------------------------------------------------------------
// Kernel 3: flash attention.  grid = (32 q-blocks of 64, 16 heads, 4 batch)
// block = 128 thr (4 waves); wave owns a 16-row q tile; key blocks of 32.
// K and V^T fragments are loaded directly from global (L2-resident, contiguous
// per lane); LDS used only to restage P from C-layout to A-layout (in-wave).
// ---------------------------------------------------------------------------
#define P_LD  40     // 32 + 8 pad (80B rows: 16B aligned, conflict-free)
__global__ __launch_bounds__(128) void attn_kernel(
    const bf16_t* __restrict__ q, const bf16_t* __restrict__ k,
    const bf16_t* __restrict__ vT, bf16_t* __restrict__ ctx)
{
    __shared__ bf16_t ps[4][16 * P_LD];

    const int qt   = blockIdx.x;
    const int h    = blockIdx.y;
    const int b    = blockIdx.z;
    const int wid  = threadIdx.x >> 5;
    const int lane = threadIdx.x & 31;
    const int crow = lane >> 4;
    const int ccol = lane & 15;
    const int bh   = b * NHEADS + h;

    const bf16_t* qb  = q  + ((size_t)bh * SEQ + qt * 64 + wid * 16) * MHEAD;
    const bf16_t* kgb = k  + (size_t)bh * SEQ * MHEAD;      // [s, m]
    const bf16_t* vgb = vT + (size_t)bh * MHEAD * SEQ;      // [m, s]

    // q fragments (K = m = 64 -> two A frags); q is pre-scaled by 1/sqrt(m)
    v16bf aq0 = load_a_frag(qb, MHEAD);
    v16bf aq1 = load_a_frag(qb + 32, MHEAD);

    v8f acc0 = {}, acc1 = {}, acc2 = {}, acc3 = {};
    float mrow[8], lrow[8];
#pragma unroll
    for (int r = 0; r < 8; ++r) { mrow[r] = -1e30f; lrow[r] = 0.0f; }

    bf16_t* myp = &ps[wid][0];

    for (int t0 = 0; t0 < SEQ; t0 += 32) {
        // prefetch next key/value block into cache (global_prefetch_b8)
        if (t0 + 32 < SEQ) {
            __builtin_prefetch(kgb + (t0 + 32) * MHEAD + lane * 64, 0, 0);
            __builtin_prefetch(vgb + (size_t)lane * SEQ + t0 + 32, 0, 0);
        }

        // scores S = q . K^T for 32 keys, as two 16x16 tiles.
        // K is [s,m]: column n of B = key row t0+c*16+n; contiguous over m.
        v8f s0 = {}, s1 = {};
        s0 = wmma_bf16(aq0, load_b_frag_T(kgb + (t0)      * MHEAD, MHEAD, 0),  s0);
        s0 = wmma_bf16(aq1, load_b_frag_T(kgb + (t0)      * MHEAD, MHEAD, 32), s0);
        s1 = wmma_bf16(aq0, load_b_frag_T(kgb + (t0 + 16) * MHEAD, MHEAD, 0),  s1);
        s1 = wmma_bf16(aq1, load_b_frag_T(kgb + (t0 + 16) * MHEAD, MHEAD, 32), s1);

        // online softmax over the 32 columns; C layout row = r + 8*crow
#pragma unroll
        for (int r = 0; r < 8; ++r) {
            float mx = fmaxf(s0[r], s1[r]);
            mx = fmaxf(mx, __shfl_xor(mx, 1));
            mx = fmaxf(mx, __shfl_xor(mx, 2));
            mx = fmaxf(mx, __shfl_xor(mx, 4));
            mx = fmaxf(mx, __shfl_xor(mx, 8));
            float mnew = fmaxf(mrow[r], mx);
            float p0 = __expf(s0[r] - mnew);
            float p1 = __expf(s1[r] - mnew);
            float corr = __expf(mrow[r] - mnew);
            float sum = p0 + p1;
            sum += __shfl_xor(sum, 1);
            sum += __shfl_xor(sum, 2);
            sum += __shfl_xor(sum, 4);
            sum += __shfl_xor(sum, 8);
            lrow[r] = lrow[r] * corr + sum;
            mrow[r] = mnew;
            acc0[r] *= corr; acc1[r] *= corr; acc2[r] *= corr; acc3[r] *= corr;
            int prow = r + crow * 8;
            myp[prow * P_LD + ccol]      = (bf16_t)p0;
            myp[prow * P_LD + ccol + 16] = (bf16_t)p1;
        }
        // LDS ops within a wave are in-order, but be explicit:
        asm volatile("s_wait_dscnt 0" ::: "memory");

        // re-load P (16x32) in A-fragment layout, then P . V (4 n-tiles).
        // V^T is [m,s]: column n of B = m index nt0+n; contiguous over t (=K).
        v16bf ap = load_a_frag(myp, P_LD);
        acc0 = wmma_bf16(ap, load_b_frag_T(vgb + (size_t) 0 * SEQ + t0, SEQ, 0), acc0);
        acc1 = wmma_bf16(ap, load_b_frag_T(vgb + (size_t)16 * SEQ + t0, SEQ, 0), acc1);
        acc2 = wmma_bf16(ap, load_b_frag_T(vgb + (size_t)32 * SEQ + t0, SEQ, 0), acc2);
        acc3 = wmma_bf16(ap, load_b_frag_T(vgb + (size_t)48 * SEQ + t0, SEQ, 0), acc3);
    }

    // normalize and store ctx (bf16)
    bf16_t* cdst = ctx + ((size_t)bh * SEQ + qt * 64 + wid * 16) * MHEAD;
#pragma unroll
    for (int r = 0; r < 8; ++r) {
        float inv = 1.0f / lrow[r];
        int row = r + crow * 8;
        cdst[row * MHEAD + ccol]      = (bf16_t)(acc0[r] * inv);
        cdst[row * MHEAD + ccol + 16] = (bf16_t)(acc1[r] * inv);
        cdst[row * MHEAD + ccol + 32] = (bf16_t)(acc2[r] * inv);
        cdst[row * MHEAD + ccol + 48] = (bf16_t)(acc3[r] * inv);
    }
}

// ---------------------------------------------------------------------------
// Kernel 4: output projection, sum over heads.
// grid = (128 s-tiles, 16 d-groups, 4 batch), 128 thr; wave -> one 16-wide d tile.
// W_o^T is [h,d,m] (N-major over d) -> contiguous B fragments.
// ---------------------------------------------------------------------------
__global__ __launch_bounds__(128) void oproj_kernel(
    const bf16_t* __restrict__ ctx, const bf16_t* __restrict__ woT,
    float* __restrict__ out)
{
    const int s0   = blockIdx.x * 16;
    const int dgrp = blockIdx.y;
    const int b    = blockIdx.z;
    const int wid  = threadIdx.x >> 5;
    const int lane = threadIdx.x & 31;
    const int dt   = dgrp * 4 + wid;            // 0..63 tiles of 16 over d=1024

    v8f acc = {};
    for (int h = 0; h < NHEADS; ++h) {
        const bf16_t* cb = ctx + ((size_t)(b * NHEADS + h) * SEQ + s0) * MHEAD;
        v16bf a0 = load_a_frag(cb, MHEAD);
        v16bf a1 = load_a_frag(cb + 32, MHEAD);
        // W_o^T layout [h,d,m]: B[k][n] = woT[(h*1024 + dt*16 + n)*64 + kofs + k]
        const bf16_t* wb = woT + (size_t)(h * DMODEL + dt * 16) * MHEAD;
        acc = wmma_bf16(a0, load_b_frag_T(wb, MHEAD, 0),  acc);
        acc = wmma_bf16(a1, load_b_frag_T(wb, MHEAD, 32), acc);
    }
    const int crow = lane >> 4, ccol = lane & 15;
    float* dst = out + ((size_t)b * SEQ + s0) * DMODEL + dt * 16;
#pragma unroll
    for (int r = 0; r < 8; ++r)
        dst[(r + crow * 8) * DMODEL + ccol] = acc[r];
}

// ---------------------------------------------------------------------------
extern "C" void kernel_launch(void* const* d_in, const int* in_sizes, int n_in,
                              void* d_out, int out_size, void* d_ws, size_t ws_size,
                              hipStream_t stream) {
    (void)in_sizes; (void)n_in; (void)out_size; (void)ws_size;
    const float* x  = (const float*)d_in[0];
    const float* Wq = (const float*)d_in[1];
    const float* Wk = (const float*)d_in[2];
    const float* Wv = (const float*)d_in[3];
    const float* Wo = (const float*)d_in[4];

    char* ws = (char*)d_ws;
    bf16_t* xh  = (bf16_t*)(ws + OFF_XH);
    bf16_t* wqT = (bf16_t*)(ws + OFF_WQ);
    bf16_t* wkT = (bf16_t*)(ws + OFF_WK);
    bf16_t* wvT = (bf16_t*)(ws + OFF_WV);
    bf16_t* woT = (bf16_t*)(ws + OFF_WO);
    bf16_t* q   = (bf16_t*)(ws + OFF_Q);
    bf16_t* k   = (bf16_t*)(ws + OFF_K);
    bf16_t* vT  = (bf16_t*)(ws + OFF_V);
    bf16_t* ctx = (bf16_t*)(ws + OFF_CTX);

    const int nx = BATCH * SEQ * DMODEL;        // 8,388,608
    const int nw = NHEADS * DMODEL * MHEAD;     // 1,048,576
    cvt_f32_bf16<<<nx / 256, 256, 0, stream>>>(x, xh, nx);
    cvt_tr_qkvw<<<nw / 256, 256, 0, stream>>>(Wq, wqT);
    cvt_tr_qkvw<<<nw / 256, 256, 0, stream>>>(Wk, wkT);
    cvt_tr_qkvw<<<nw / 256, 256, 0, stream>>>(Wv, wvT);
    cvt_tr_ow  <<<nw / 256, 256, 0, stream>>>(Wo, woT);

    qkv_kernel<<<dim3(SEQ / 16, BATCH), 384, 0, stream>>>(xh, wqT, wkT, wvT, q, k, vT);
    attn_kernel<<<dim3(SEQ / 64, NHEADS, BATCH), 128, 0, stream>>>(q, k, vT, ctx);
    oproj_kernel<<<dim3(SEQ / 16, DMODEL / 64, BATCH), 128, 0, stream>>>(ctx, woT, (float*)d_out);
}
